// ObjectTransformer_23407571763397
// MI455X (gfx1250) — matrix operand order repaired and checked by reading, not compiled, not run on hardware
//
#include <hip/hip_runtime.h>
#include <hip/hip_bf16.h>
#include <cstdint>
#include <cstddef>

// ---------------- problem constants ----------------
#define EMBED   768
#define HEADS   12
#define DEPTH   12
#define FEAT    2048
#define BATCH   256
#define FRAMES  4
#define LTOK    20
#define NTOK    81                    // 1 + FRAMES*LTOK
#define MROWS   (BATCH*NTOK)          // 20736 = 128*162
#define MSPAT   (BATCH*FRAMES*LTOK)   // 20480 = 128*160
#define XROW    2054
#define HDIM    64                    // EMBED/HEADS

typedef __attribute__((ext_vector_type(16))) __bf16 v16bf;
typedef __attribute__((ext_vector_type(8)))  float  v8f;

union Frag16 {
  unsigned short s[16];
  uint4          u[2];
  v16bf          v;
};

__device__ __forceinline__ unsigned short f2bf(float f) {
  unsigned int u = __float_as_uint(f);
  u += 0x7FFFu + ((u >> 16) & 1u);          // round-to-nearest-even
  return (unsigned short)(u >> 16);
}
__device__ __forceinline__ float bf2f(unsigned short s) {
  return __uint_as_float(((unsigned int)s) << 16);
}
__device__ __forceinline__ float gelu_exact(float x) {
  return 0.5f * x * (1.0f + erff(x * 0.70710678118654752f));
}

// ---------------- elementwise f32 -> bf16 ----------------
__global__ void __launch_bounds__(256)
k_f32_to_bf16(const float* __restrict__ src, unsigned short* __restrict__ dst, size_t n) {
  size_t i = (size_t)blockIdx.x * blockDim.x + threadIdx.x;
  size_t stride = (size_t)gridDim.x * blockDim.x;
  for (; i < n; i += stride) dst[i] = f2bf(src[i]);
}

// ---------------- generic WMMA GEMM: C[M,N] = A[M,K](bf16) * W[N,K](bf16)^T ----------------
// 8 waves / block arranged 4(M) x 2(N); each wave owns a 32x32 micro-tile:
// 2 A-frags x 2 B-frags -> 4 WMMAs per 32-K step (2x register reuse on both operands).
// Block tile = 128 x 64.
__global__ void __launch_bounds__(256)
k_gemm_bf16(const unsigned short* __restrict__ A, int lda,
            const unsigned short* __restrict__ W, int K,
            const float* __restrict__ bias,
            const float* __restrict__ resid,
            float* __restrict__ Cf,
            unsigned short* __restrict__ Cb,
            int M, int N, int act) {
  const int lane = threadIdx.x & 31;
  const int wave = threadIdx.x >> 5;
  const int wm = wave & 3, wn = wave >> 2;
  const int tileM = (blockIdx.y * 4 + wm) * 32;
  const int tileN = (blockIdx.x * 2 + wn) * 32;
  if (tileM >= M || tileN >= N) return;

  const int half = lane >> 4;      // K-halves per CDNA5 16-bit fragment layout
  const int l15  = lane & 15;

  const unsigned short* Ap0 = A + (size_t)(tileM + l15)      * lda + half * 8;
  const unsigned short* Ap1 = A + (size_t)(tileM + 16 + l15) * lda + half * 8;
  const unsigned short* Wp0 = W + (size_t)(tileN + l15)      * K   + half * 8;
  const unsigned short* Wp1 = W + (size_t)(tileN + 16 + l15) * K   + half * 8;

  v8f acc00 = {}, acc01 = {}, acc10 = {}, acc11 = {};
  for (int k0 = 0; k0 < K; k0 += 32) {
    Frag16 a0, a1, b0, b1;
    a0.u[0] = *(const uint4*)(Ap0 + k0);        // K = half*8 + 0..7
    a0.u[1] = *(const uint4*)(Ap0 + k0 + 16);   // K = 16 + half*8 + 0..7
    a1.u[0] = *(const uint4*)(Ap1 + k0);
    a1.u[1] = *(const uint4*)(Ap1 + k0 + 16);
    b0.u[0] = *(const uint4*)(Wp0 + k0);        // K = half*8 + 0..7   (col = l15)
    b0.u[1] = *(const uint4*)(Wp0 + k0 + 16);   // K = 16 + half*8 + 0..7
    b1.u[0] = *(const uint4*)(Wp1 + k0);
    b1.u[1] = *(const uint4*)(Wp1 + k0 + 16);
    __builtin_prefetch(Ap0 + k0 + 128, 0, 1);
    __builtin_prefetch(Ap1 + k0 + 128, 0, 1);
    __builtin_prefetch(Wp0 + k0 + 128, 0, 1);
    __builtin_prefetch(Wp1 + k0 + 128, 0, 1);
    acc00 = __builtin_amdgcn_wmma_f32_16x16x32_bf16(false, a0.v, false, b0.v, (short)0, acc00, false, false);
    acc01 = __builtin_amdgcn_wmma_f32_16x16x32_bf16(false, a0.v, false, b1.v, (short)0, acc01, false, false);
    acc10 = __builtin_amdgcn_wmma_f32_16x16x32_bf16(false, a1.v, false, b0.v, (short)0, acc10, false, false);
    acc11 = __builtin_amdgcn_wmma_f32_16x16x32_bf16(false, a1.v, false, b1.v, (short)0, acc11, false, false);
  }

  auto store_tile = [&](const v8f& acc, int tm, int tn) {
    const float bn = bias ? bias[tn + l15] : 0.0f;
#pragma unroll
    for (int r = 0; r < 8; ++r) {
      int m = tm + r + half * 8;
      int n = tn + l15;
      float v = acc[r] + bn;
      if (act == 1) v = gelu_exact(v);
      if (resid) v += resid[(size_t)m * N + n];
      if (Cf) Cf[(size_t)m * N + n] = v;
      if (Cb) Cb[(size_t)m * N + n] = f2bf(v);
    }
  };
  store_tile(acc00, tileM,      tileN);
  store_tile(acc01, tileM,      tileN + 16);
  store_tile(acc10, tileM + 16, tileN);
  store_tile(acc11, tileM + 16, tileN + 16);
}

// ---------------- embedding GEMM: A is f32 with row stride lda (reads x[..., :2048]) ---------
// Same 32x32-per-wave structure; output rows remapped m -> b*81 + 1 + within (skip cls slot).
__global__ void __launch_bounds__(256)
k_gemm_f32a(const float* __restrict__ A, int lda,
            const unsigned short* __restrict__ W, int K,
            float* __restrict__ Cf, int M, int N) {
  const int lane = threadIdx.x & 31;
  const int wave = threadIdx.x >> 5;
  const int wm = wave & 3, wn = wave >> 2;
  const int tileM = (blockIdx.y * 4 + wm) * 32;
  const int tileN = (blockIdx.x * 2 + wn) * 32;
  if (tileM >= M || tileN >= N) return;

  const int half = lane >> 4;
  const int l15  = lane & 15;

  const float*          Ap0 = A + (size_t)(tileM + l15)      * lda + half * 8;
  const float*          Ap1 = A + (size_t)(tileM + 16 + l15) * lda + half * 8;
  const unsigned short* Wp0 = W + (size_t)(tileN + l15)      * K   + half * 8;
  const unsigned short* Wp1 = W + (size_t)(tileN + 16 + l15) * K   + half * 8;

  auto load_a = [&](const float* p, Frag16& a) {
#pragma unroll
    for (int i = 0; i < 4; ++i) {                    // K = half*8 + 0..7
      float2 t = *(const float2*)(p + 2 * i);
      a.s[2 * i]     = f2bf(t.x);
      a.s[2 * i + 1] = f2bf(t.y);
    }
#pragma unroll
    for (int i = 0; i < 4; ++i) {                    // K = 16 + half*8 + 0..7
      float2 t = *(const float2*)(p + 16 + 2 * i);
      a.s[8 + 2 * i]     = f2bf(t.x);
      a.s[8 + 2 * i + 1] = f2bf(t.y);
    }
  };

  v8f acc00 = {}, acc01 = {}, acc10 = {}, acc11 = {};
  for (int k0 = 0; k0 < K; k0 += 32) {
    Frag16 a0, a1, b0, b1;
    load_a(Ap0 + k0, a0);
    load_a(Ap1 + k0, a1);
    b0.u[0] = *(const uint4*)(Wp0 + k0);
    b0.u[1] = *(const uint4*)(Wp0 + k0 + 16);
    b1.u[0] = *(const uint4*)(Wp1 + k0);
    b1.u[1] = *(const uint4*)(Wp1 + k0 + 16);
    __builtin_prefetch(Ap0 + k0 + 64, 0, 1);
    __builtin_prefetch(Ap1 + k0 + 64, 0, 1);
    __builtin_prefetch(Wp0 + k0 + 128, 0, 1);
    __builtin_prefetch(Wp1 + k0 + 128, 0, 1);
    acc00 = __builtin_amdgcn_wmma_f32_16x16x32_bf16(false, a0.v, false, b0.v, (short)0, acc00, false, false);
    acc01 = __builtin_amdgcn_wmma_f32_16x16x32_bf16(false, a0.v, false, b1.v, (short)0, acc01, false, false);
    acc10 = __builtin_amdgcn_wmma_f32_16x16x32_bf16(false, a1.v, false, b0.v, (short)0, acc10, false, false);
    acc11 = __builtin_amdgcn_wmma_f32_16x16x32_bf16(false, a1.v, false, b1.v, (short)0, acc11, false, false);
  }

  auto store_tile = [&](const v8f& acc, int tm, int tn) {
#pragma unroll
    for (int r = 0; r < 8; ++r) {
      int m = tm + r + half * 8;
      int n = tn + l15;
      int mo = m + m / (FRAMES * LTOK) + 1;          // b*81 + 1 + within
      Cf[(size_t)mo * N + n] = acc[r];
    }
  };
  store_tile(acc00, tileM,      tileN);
  store_tile(acc01, tileM,      tileN + 16);
  store_tile(acc10, tileM + 16, tileN);
  store_tile(acc11, tileM + 16, tileN + 16);
}

// ---------------- LayerNorm: wave per row, output bf16 ----------------
__global__ void __launch_bounds__(256)
k_layernorm(const float* __restrict__ x, const float* __restrict__ w,
            const float* __restrict__ b, unsigned short* __restrict__ out, int M) {
  const int lane = threadIdx.x & 31;
  const int wave = threadIdx.x >> 5;
  const int row  = blockIdx.x * 8 + wave;
  if (row >= M) return;
  const float* xr = x + (size_t)row * EMBED;
  float vals[EMBED / 32];
  float s = 0.0f;
#pragma unroll
  for (int i = 0; i < EMBED / 32; ++i) { vals[i] = xr[lane + 32 * i]; s += vals[i]; }
#pragma unroll
  for (int off = 16; off; off >>= 1) s += __shfl_xor(s, off, 32);
  const float mean = s * (1.0f / EMBED);
  float v2 = 0.0f;
#pragma unroll
  for (int i = 0; i < EMBED / 32; ++i) { float d = vals[i] - mean; v2 += d * d; }
#pragma unroll
  for (int off = 16; off; off >>= 1) v2 += __shfl_xor(v2, off, 32);
  const float rs = rsqrtf(v2 * (1.0f / EMBED) + 1e-6f);
  unsigned short* o = out + (size_t)row * EMBED;
#pragma unroll
  for (int i = 0; i < EMBED / 32; ++i) {
    int c = lane + 32 * i;
    o[c] = f2bf((vals[i] - mean) * rs * w[c] + b[c]);
  }
}

// ---------------- variable attention (cls: 81 keys; spatial: cls + 20 frame keys) -----------
// qkv: bf16 (MROWS x 2304); out: bf16 (MROWS x 768), head-major concat.
__global__ void __launch_bounds__(256)
k_attention(const unsigned short* __restrict__ qkv, const float* __restrict__ mbias,
            unsigned short* __restrict__ out) {
  const int bh = blockIdx.x;
  const int b = bh / HEADS, h = bh % HEADS;
  const int lane = threadIdx.x & 31;
  const int wave = threadIdx.x >> 5;
  const size_t base = (size_t)b * NTOK;

  for (int t = wave; t < NTOK; t += 8) {
    const unsigned short* qr = qkv + (base + t) * (3 * EMBED) + h * HDIM;
    const float scale = 0.125f;                       // 64^-0.5
    float q0 = bf2f(qr[lane * 2])     * scale;
    float q1 = bf2f(qr[lane * 2 + 1]) * scale;

    float mx = -1e30f, den = 0.0f, o0 = 0.0f, o1 = 0.0f;
    auto step = [&](int j) {
      const unsigned short* kr = qkv + (base + j) * (3 * EMBED) + EMBED + h * HDIM;
      float s = q0 * bf2f(kr[lane * 2]) + q1 * bf2f(kr[lane * 2 + 1]);
#pragma unroll
      for (int off = 16; off; off >>= 1) s += __shfl_xor(s, off, 32);
      s += mbias[base + j];
      float mn = fmaxf(mx, s);
      float corr = __expf(mx - mn);
      float p    = __expf(s - mn);
      const unsigned short* vr = qkv + (base + j) * (3 * EMBED) + 2 * EMBED + h * HDIM;
      o0 = o0 * corr + p * bf2f(vr[lane * 2]);
      o1 = o1 * corr + p * bf2f(vr[lane * 2 + 1]);
      den = den * corr + p;
      mx = mn;
    };

    if (t == 0) {
      for (int j = 0; j < NTOK; ++j) step(j);
    } else {
      const int f = (t - 1) / LTOK;
      step(0);                                        // cls key
      const int j0 = 1 + f * LTOK;
      for (int j = j0; j < j0 + LTOK; ++j) step(j);
    }
    const float inv = 1.0f / den;
    unsigned short* orow = out + (base + t) * EMBED + h * HDIM;
    orow[lane * 2]     = f2bf(o0 * inv);
    orow[lane * 2 + 1] = f2bf(o1 * inv);
  }
}

// ---------------- mask bias: 0 for valid, -100 for padded; also written to output tail ------
__global__ void __launch_bounds__(256)
k_mbias(const int* __restrict__ x_mask, float* __restrict__ mb, float* __restrict__ tail) {
  int idx = blockIdx.x * blockDim.x + threadIdx.x;
  if (idx >= MROWS) return;
  int b = idx / NTOK, j = idx % NTOK;
  float v = 0.0f;
  if (j > 0) v = ((float)x_mask[b * (FRAMES * LTOK) + (j - 1)] - 1.0f) * 100.0f;
  mb[idx] = v;
  tail[idx] = v;
}

// ---------------- cls row init: cls_token + cls_pos[0,0,:] ----------------
__global__ void __launch_bounds__(256)
k_cls(const float* __restrict__ cls_token, const float* __restrict__ cls_pos,
      float* __restrict__ tok) {
  int idx = blockIdx.x * blockDim.x + threadIdx.x;
  if (idx >= BATCH * EMBED) return;
  int b = idx / EMBED, n = idx % EMBED;
  tok[(size_t)b * NTOK * EMBED + n] = cls_token[n] + cls_pos[n];
}

// ---------------- embedding fixup: + obj_b + pos_b + x[...,2048:]@pos_W^T + temporal --------
__global__ void __launch_bounds__(256)
k_embed_fixup(const float* __restrict__ x, const float* __restrict__ obj_b,
              const float* __restrict__ pos_W, const float* __restrict__ pos_b,
              const float* __restrict__ temporal, float* __restrict__ tok) {
  size_t idx = (size_t)blockIdx.x * blockDim.x + threadIdx.x;
  if (idx >= (size_t)MSPAT * EMBED) return;
  int m = (int)(idx / EMBED), n = (int)(idx % EMBED);
  int b = m / (FRAMES * LTOK), within = m % (FRAMES * LTOK), f = within / LTOK;
  size_t o = ((size_t)b * NTOK + 1 + within) * EMBED + n;
  const float* xp = x + (size_t)m * XROW + FEAT;
  float v = tok[o] + obj_b[n] + pos_b[n] + temporal[f * EMBED + n];
#pragma unroll
  for (int k = 0; k < 6; ++k) v += xp[k] * pos_W[n * 6 + k];
  tok[o] = v;
}

// ======================= host orchestration =======================
extern "C" void kernel_launch(void* const* d_in, const int* in_sizes, int n_in,
                              void* d_out, int out_size, void* d_ws, size_t ws_size,
                              hipStream_t stream) {
  const float* x         = (const float*)d_in[0];
  const int*   x_mask    = (const int*)d_in[1];
  const float* obj_W     = (const float*)d_in[2];
  const float* obj_b     = (const float*)d_in[3];
  const float* pos_W     = (const float*)d_in[4];
  const float* pos_b     = (const float*)d_in[5];
  const float* cls_token = (const float*)d_in[6];
  const float* cls_pos   = (const float*)d_in[7];
  const float* temporal  = (const float*)d_in[8];
  const float* ln1_w     = (const float*)d_in[9];
  const float* ln1_b     = (const float*)d_in[10];
  const float* ln2_w     = (const float*)d_in[11];
  const float* ln2_b     = (const float*)d_in[12];
  const float* qkv_W     = (const float*)d_in[13];
  const float* qkv_b     = (const float*)d_in[14];
  const float* ap_W      = (const float*)d_in[15];
  const float* ap_b      = (const float*)d_in[16];
  const float* fc1_W     = (const float*)d_in[17];
  const float* fc1_b     = (const float*)d_in[18];
  const float* fc2_W     = (const float*)d_in[19];
  const float* fc2_b     = (const float*)d_in[20];
  const float* proj_W    = (const float*)d_in[21];

  // ---- workspace carve-out (256B aligned) ----
  char* ws = (char*)d_ws;
  size_t off = 0;
  auto take = [&](size_t elems, size_t esize) -> void* {
    void* p = ws + off;
    off += ((elems * esize + 255) & ~(size_t)255);
    return p;
  };
  unsigned short* wobj = (unsigned short*)take((size_t)EMBED * FEAT, 2);
  unsigned short* wqkv = (unsigned short*)take((size_t)DEPTH * 3 * EMBED * EMBED, 2);
  unsigned short* wap  = (unsigned short*)take((size_t)DEPTH * EMBED * EMBED, 2);
  unsigned short* wfc1 = (unsigned short*)take((size_t)DEPTH * 4 * EMBED * EMBED, 2);
  unsigned short* wfc2 = (unsigned short*)take((size_t)DEPTH * 4 * EMBED * EMBED, 2);
  unsigned short* wprj = (unsigned short*)take((size_t)256 * EMBED, 2);
  float*          tok  = (float*)take((size_t)MROWS * EMBED, 4);
  unsigned short* lnbf = (unsigned short*)take((size_t)MROWS * EMBED, 2);
  unsigned short* atbf = (unsigned short*)take((size_t)MROWS * EMBED, 2);
  unsigned short* qkvb = (unsigned short*)take((size_t)MROWS * 3 * EMBED, 2);
  unsigned short* hidb = (unsigned short*)take((size_t)MROWS * 4 * EMBED, 2);
  float*          mb   = (float*)take((size_t)MROWS, 4);

  // ---- weight conversion to bf16 ----
  dim3 cb(256);
  k_f32_to_bf16<<<2048, cb, 0, stream>>>(obj_W, wobj, (size_t)EMBED * FEAT);
  k_f32_to_bf16<<<2048, cb, 0, stream>>>(qkv_W, wqkv, (size_t)DEPTH * 3 * EMBED * EMBED);
  k_f32_to_bf16<<<2048, cb, 0, stream>>>(ap_W,  wap,  (size_t)DEPTH * EMBED * EMBED);
  k_f32_to_bf16<<<2048, cb, 0, stream>>>(fc1_W, wfc1, (size_t)DEPTH * 4 * EMBED * EMBED);
  k_f32_to_bf16<<<2048, cb, 0, stream>>>(fc2_W, wfc2, (size_t)DEPTH * 4 * EMBED * EMBED);
  k_f32_to_bf16<<<2048, cb, 0, stream>>>(proj_W, wprj, (size_t)256 * EMBED);

  // ---- embedding: tok[spatial] = x[...,:2048] @ obj_W^T (remapped rows) ----
  k_gemm_f32a<<<dim3(EMBED / 64, MSPAT / 128), 256, 0, stream>>>(
      x, XROW, wobj, FEAT, tok, MSPAT, EMBED);
  k_embed_fixup<<<(unsigned)(((size_t)MSPAT * EMBED + 255) / 256), 256, 0, stream>>>(
      x, obj_b, pos_W, pos_b, temporal, tok);
  k_cls<<<(BATCH * EMBED + 255) / 256, 256, 0, stream>>>(cls_token, cls_pos, tok);
  float* out_tail = (float*)d_out + (size_t)MROWS * 256;
  k_mbias<<<(MROWS + 255) / 256, 256, 0, stream>>>(x_mask, mb, out_tail);

  // ---- 12 transformer layers ----
  for (int l = 0; l < DEPTH; ++l) {
    const unsigned short* wq = wqkv + (size_t)l * 3 * EMBED * EMBED;
    const unsigned short* wa = wap  + (size_t)l * EMBED * EMBED;
    const unsigned short* w1 = wfc1 + (size_t)l * 4 * EMBED * EMBED;
    const unsigned short* w2 = wfc2 + (size_t)l * 4 * EMBED * EMBED;

    k_layernorm<<<MROWS / 8, 256, 0, stream>>>(tok, ln1_w + l * EMBED, ln1_b + l * EMBED,
                                               lnbf, MROWS);
    // qkv: (MROWS x 768) x (2304 x 768)^T -> bf16
    k_gemm_bf16<<<dim3(3 * EMBED / 64, MROWS / 128), 256, 0, stream>>>(
        lnbf, EMBED, wq, EMBED, qkv_b + (size_t)l * 3 * EMBED,
        nullptr, nullptr, qkvb, MROWS, 3 * EMBED, 0);
    k_attention<<<BATCH * HEADS, 256, 0, stream>>>(qkvb, mb, atbf);
    // attn proj + residual -> tok (f32, in place)
    k_gemm_bf16<<<dim3(EMBED / 64, MROWS / 128), 256, 0, stream>>>(
        atbf, EMBED, wa, EMBED, ap_b + (size_t)l * EMBED,
        tok, tok, nullptr, MROWS, EMBED, 0);
    k_layernorm<<<MROWS / 8, 256, 0, stream>>>(tok, ln2_w + l * EMBED, ln2_b + l * EMBED,
                                               lnbf, MROWS);
    // fc1 + exact GELU -> bf16 hidden
    k_gemm_bf16<<<dim3(4 * EMBED / 64, MROWS / 128), 256, 0, stream>>>(
        lnbf, EMBED, w1, EMBED, fc1_b + (size_t)l * 4 * EMBED,
        nullptr, nullptr, hidb, MROWS, 4 * EMBED, 1);
    // fc2 + residual -> tok (f32, in place)
    k_gemm_bf16<<<dim3(EMBED / 64, MROWS / 128), 256, 0, stream>>>(
        hidb, 4 * EMBED, w2, 4 * EMBED, fc2_b + (size_t)l * EMBED,
        tok, tok, nullptr, MROWS, EMBED, 0);
  }

  // ---- final projection: (MROWS x 768) x (256 x 768)^T -> d_out ----
  k_f32_to_bf16<<<2048, cb, 0, stream>>>(tok, lnbf, (size_t)MROWS * EMBED);
  k_gemm_bf16<<<dim3(256 / 64, MROWS / 128), 256, 0, stream>>>(
      lnbf, EMBED, wprj, EMBED, nullptr,
      nullptr, (float*)d_out, nullptr, MROWS, 256, 0);
}
